// CrossModalMoELayer_48103633715323
// MI455X (gfx1250) — compile-verified, matrix operand order checked by reading, not run on hardware
//
#include <hip/hip_runtime.h>
#include <hip/hip_bf16.h>

#define BQ   8
#define TLEN 256
#define NHH  16
#define HDIM 1024
#define HD   64
#define FDIM 4096
#define NE   8

typedef _Float16 v16h __attribute__((ext_vector_type(16)));
typedef _Float16 h8   __attribute__((ext_vector_type(8)));
typedef float    v8f  __attribute__((ext_vector_type(8)));
typedef int      v4i  __attribute__((ext_vector_type(4)));

#define WMMA_F16(a, b, c) \
  __builtin_amdgcn_wmma_f32_16x16x32_f16(false, (a), false, (b), (short)0, (c), false, false)

// --- gfx1250 async global->LDS path (ASYNCcnt), with safe fallback ----------
#if __has_builtin(__builtin_amdgcn_global_load_async_to_lds_b128)
#define HAVE_ASYNC_LDS 1
typedef __attribute__((address_space(1))) v4i* gv4p;   // global v4i*
typedef __attribute__((address_space(3))) v4i* lv4p;   // LDS v4i*
__device__ __forceinline__ void wait_async_n(bool keep4) {
#if __has_builtin(__builtin_amdgcn_s_wait_asynccnt)
  if (keep4) __builtin_amdgcn_s_wait_asynccnt(4);
  else       __builtin_amdgcn_s_wait_asynccnt(0);
#else
  if (keep4) asm volatile("s_wait_asynccnt 4" ::: "memory");
  else       asm volatile("s_wait_asynccnt 0" ::: "memory");
#endif
}
#else
#define HAVE_ASYNC_LDS 0
#endif

// Load a 16x32 f16 fragment (A layout, or B layout from an [N,K] row-major tile).
// ISA 16-bit A layout: lanes 0-15 -> M=lane, K in {0..7,16..23}; lanes 16-31 -> M=lane-16, K in {8..15,24..31}.
__device__ __forceinline__ v16h ld_frag(const _Float16* p, int stride) {
  int lane = threadIdx.x & 31;
  int r = lane & 15, hi = lane >> 4;
  const _Float16* rowp = p + r * stride + 8 * hi;
  union { v16h v; h8 h[2]; } u;
  u.h[0] = *(const h8*)(rowp);        // K = 8*hi .. 8*hi+7
  u.h[1] = *(const h8*)(rowp + 16);   // K = 16+8*hi .. 16+8*hi+7
  return u.v;
}

// Stage a 128x32 f16 A-tile and B-tile into LDS (each thread moves 2x16B per
// matrix). Async variant issues 4 GLOBAL_LOAD_ASYNC_TO_LDS_B128 per thread.
__device__ __forceinline__ void stage_tile(_Float16* dstA, _Float16* dstB,
                                           const _Float16* Ab, const _Float16* Bb,
                                           int K, int k0) {
  const int t = threadIdx.x;
#pragma unroll
  for (int i = 0; i < 2; i++) {
    int idx = t + i * 256;                 // 512 chunks of 8 halves per matrix
    int row = idx >> 2, ch = (idx & 3) * 8;
#if HAVE_ASYNC_LDS
    __builtin_amdgcn_global_load_async_to_lds_b128(
        (gv4p)&Ab[(size_t)row * K + k0 + ch], (lv4p)&dstA[row * 40 + ch], 0, 0);
    __builtin_amdgcn_global_load_async_to_lds_b128(
        (gv4p)&Bb[(size_t)row * K + k0 + ch], (lv4p)&dstB[row * 40 + ch], 0, 0);
#else
    *(h8*)&dstA[row * 40 + ch] = *(const h8*)&Ab[(size_t)row * K + k0 + ch];
    *(h8*)&dstB[row * 40 + ch] = *(const h8*)&Bb[(size_t)row * K + k0 + ch];
#endif
  }
}

// ---------------------------------------------------------------------------
// Generic WMMA GEMM: C = A[M,K] * B[N,K]^T (+ epilogue). Macro tile 128x128,
// 8 waves (2 M-waves x 4 N-waves), each wave 64x32 = 4x2 fragments.
// LDS double-buffered: tile k+1 is async-prefetched while tile k feeds WMMAs.
// MODE 0: f16 out = acc + bias
// MODE 1: f32 out = acc + bias + extra (residual)
// MODE 2: f16 out = gelu(acc + bias)       (exact erf GELU)
// MODE 3: f32 out += extra[row*NE+e_idx] * (acc + bias)   (MoE accumulate)
// ---------------------------------------------------------------------------
template <int MODE>
__global__ void gemm_wmma(const _Float16* __restrict__ A, const _Float16* __restrict__ B,
                          const float* __restrict__ bias, const float* __restrict__ extra,
                          float* __restrict__ outf, _Float16* __restrict__ outh,
                          int M, int N, int K, int e_idx) {
  __shared__ _Float16 As[2][128 * 40];
  __shared__ _Float16 Bs[2][128 * 40];
  const int t = threadIdx.x;
  const int wave = t >> 5, lane = t & 31;
  const int wm = wave & 1, wn = wave >> 1;
  const int bm = blockIdx.x, bn = blockIdx.y;
  const _Float16* Ab = A + (size_t)(bm * 128) * K;
  const _Float16* Bb = B + (size_t)(bn * 128) * K;

  v8f c[4][2] = {};
  stage_tile(&As[0][0], &Bs[0][0], Ab, Bb, K, 0);
  int cur = 0;
  for (int k0 = 0; k0 < K; k0 += 32) {
    const bool pre = (k0 + 32 < K);
    if (pre) stage_tile(&As[cur ^ 1][0], &Bs[cur ^ 1][0], Ab, Bb, K, k0 + 32);
#if HAVE_ASYNC_LDS
    wait_async_n(pre);   // current tile resident; next tile stays in flight
#endif
    __syncthreads();
    v16h a[4], b[2];
#pragma unroll
    for (int i = 0; i < 4; i++) a[i] = ld_frag(&As[cur][(wm * 64 + i * 16) * 40], 40);
#pragma unroll
    for (int j = 0; j < 2; j++) b[j] = ld_frag(&Bs[cur][(wn * 32 + j * 16) * 40], 40);
#pragma unroll
    for (int i = 0; i < 4; i++)
#pragma unroll
      for (int j = 0; j < 2; j++) c[i][j] = WMMA_F16(a[i], b[j], c[i][j]);
    __syncthreads();
    cur ^= 1;
  }

  const int ln = lane & 15, hb = lane >> 4;
#pragma unroll
  for (int i = 0; i < 4; i++)
#pragma unroll
    for (int j = 0; j < 2; j++) {
      int col = bn * 128 + wn * 32 + j * 16 + ln;
      float bv = bias ? bias[col] : 0.f;
#pragma unroll
      for (int r = 0; r < 8; r++) {
        int row = bm * 128 + wm * 64 + i * 16 + r + 8 * hb;
        float v = c[i][j][r] + bv;
        size_t o = (size_t)row * N + col;
        if (MODE == 0) outh[o] = (_Float16)v;
        else if (MODE == 1) outf[o] = v + extra[o];
        else if (MODE == 2) outh[o] = (_Float16)(0.5f * v * (1.f + erff(v * 0.70710678f)));
        else if (MODE == 3) outf[o] += extra[(size_t)row * NE + e_idx] * v;
      }
    }
}

// ---------------------------------------------------------------------------
// Fused attention (flash-style, online softmax). Workgroup = (b, head, 128 q
// rows); 8 waves, each owns a 16-row Q tile. K/V tiles (32 rows) staged in LDS
// (K via async-to-LDS, V transposed), S = Q K^T and O += P V via WMMA.
// qkv layout: [B*T, 3H] f16, Q at col 0, K at col H, V at col 2H.
// ---------------------------------------------------------------------------
__global__ void attn_wmma(const _Float16* __restrict__ qkv_q,
                          const _Float16* __restrict__ qkv_kv,
                          _Float16* __restrict__ out, int Tq, int Tk) {
  __shared__ _Float16 Kt[32 * 72];       // K tile [32 rows][64 + pad]
  __shared__ _Float16 Vt[64 * 40];       // V tile transposed [64 cols][32 + pad]
  __shared__ _Float16 Pl[8 * 16 * 40];   // per-wave P staging [16][32 + pad]
  const int gid = blockIdx.x;
  const int nqb = Tq >> 7;
  const int qb = gid % nqb;
  const int h = (gid / nqb) % NHH;
  const int b = gid / (nqb * NHH);
  const int t = threadIdx.x, wave = t >> 5, lane = t & 31, ln = lane & 15, hb = lane >> 4;
  const int qrow = qb * 128 + wave * 16;

  const _Float16* qbase = qkv_q + ((size_t)(b * Tq + qrow)) * (3 * HDIM) + h * HD;
  v16h aq0 = ld_frag(qbase, 3 * HDIM);        // Q k=0..31
  v16h aq1 = ld_frag(qbase + 32, 3 * HDIM);   // Q k=32..63

  v8f o[4] = {};
  float mrow[8], lrow[8];
#pragma unroll
  for (int r = 0; r < 8; r++) { mrow[r] = -3.0e38f; lrow[r] = 0.f; }
  _Float16* pw = &Pl[wave * 640];

  for (int tk = 0; tk < Tk; tk += 32) {
    {  // cooperative K/V tile load: 256 threads x 16B each matrix
      int row = t >> 3, ch = (t & 7) * 8;
      const _Float16* kp = qkv_kv + ((size_t)(b * Tk + tk + row)) * (3 * HDIM) + HDIM + h * HD + ch;
#if HAVE_ASYNC_LDS
      __builtin_amdgcn_global_load_async_to_lds_b128((gv4p)kp, (lv4p)&Kt[row * 72 + ch], 0, 0);
#else
      *(h8*)&Kt[row * 72 + ch] = *(const h8*)kp;
#endif
      h8 vv = *(const h8*)(kp + HDIM);        // V row -> transposed store
#pragma unroll
      for (int j = 0; j < 8; j++) Vt[(ch + j) * 40 + row] = vv[j];
    }
#if HAVE_ASYNC_LDS
    wait_async_n(false);
#endif
    __syncthreads();

    v16h b00 = ld_frag(&Kt[0], 72);            // keys tk..tk+15, k 0..31
    v16h b01 = ld_frag(&Kt[32], 72);           // keys tk..tk+15, k 32..63
    v16h b10 = ld_frag(&Kt[16 * 72], 72);      // keys tk+16..tk+31
    v16h b11 = ld_frag(&Kt[16 * 72 + 32], 72);
    v8f s[2] = {};
    s[0] = WMMA_F16(aq0, b00, s[0]);
    s[0] = WMMA_F16(aq1, b01, s[0]);
    s[1] = WMMA_F16(aq0, b10, s[1]);
    s[1] = WMMA_F16(aq1, b11, s[1]);

#pragma unroll
    for (int r = 0; r < 8; r++) {
      float sc0 = s[0][r] * 0.125f, sc1 = s[1][r] * 0.125f;   // 1/sqrt(64)
      float v = fmaxf(sc0, sc1);
      v = fmaxf(v, __shfl_xor(v, 1)); v = fmaxf(v, __shfl_xor(v, 2));
      v = fmaxf(v, __shfl_xor(v, 4)); v = fmaxf(v, __shfl_xor(v, 8));
      float mnew = fmaxf(mrow[r], v);
      float alpha = __expf(mrow[r] - mnew);
      float p0 = __expf(sc0 - mnew), p1 = __expf(sc1 - mnew);
      float rs = p0 + p1;
      rs += __shfl_xor(rs, 1); rs += __shfl_xor(rs, 2);
      rs += __shfl_xor(rs, 4); rs += __shfl_xor(rs, 8);
      lrow[r] = lrow[r] * alpha + rs;
      mrow[r] = mnew;
      o[0][r] *= alpha; o[1][r] *= alpha; o[2][r] *= alpha; o[3][r] *= alpha;
      pw[(r + 8 * hb) * 40 + ln] = (_Float16)p0;        // P -> LDS (A layout staging)
      pw[(r + 8 * hb) * 40 + 16 + ln] = (_Float16)p1;
    }
    asm volatile("s_wait_dscnt 0" ::: "memory");  // own-wave DS RAW before re-read
    v16h ap = ld_frag(pw, 40);
#pragma unroll
    for (int vc = 0; vc < 4; vc++) {
      v16h bv = ld_frag(&Vt[vc * 16 * 40], 40);   // V cols vc*16.., k 0..31
      o[vc] = WMMA_F16(ap, bv, o[vc]);
    }
    __syncthreads();
  }

#pragma unroll
  for (int vc = 0; vc < 4; vc++)
#pragma unroll
    for (int r = 0; r < 8; r++) {
      int m = qrow + r + 8 * hb;
      out[((size_t)(b * Tq + m)) * HDIM + h * HD + vc * 16 + ln] = (_Float16)(o[vc][r] / lrow[r]);
    }
}

// ---------------------------------------------------------------------------
// Small fp32 helper kernels (LN, conversions, gating) — negligible FLOPs.
// ---------------------------------------------------------------------------
__global__ void ln_f16(const float* __restrict__ x, const float* __restrict__ g,
                       const float* __restrict__ bb, _Float16* __restrict__ y, int C) {
  __shared__ float s1[256], s2[256];
  int row = blockIdx.x;
  const float* xr = x + (size_t)row * C;
  float a = 0.f, b = 0.f;
  for (int i = threadIdx.x; i < C; i += 256) { float v = xr[i]; a += v; b += v * v; }
  s1[threadIdx.x] = a; s2[threadIdx.x] = b; __syncthreads();
  for (int st = 128; st > 0; st >>= 1) {
    if (threadIdx.x < st) { s1[threadIdx.x] += s1[threadIdx.x + st]; s2[threadIdx.x] += s2[threadIdx.x + st]; }
    __syncthreads();
  }
  float mean = s1[0] / C;
  float inv = rsqrtf(s2[0] / C - mean * mean + 1e-5f);
  for (int i = threadIdx.x; i < C; i += 256)
    y[(size_t)row * C + i] = (_Float16)((xr[i] - mean) * inv * g[i] + bb[i]);
}

__global__ void cvt_f16(const float* __restrict__ x, _Float16* __restrict__ y, int n) {
  int i = blockIdx.x * 256 + threadIdx.x;
  if (i < n) y[i] = (_Float16)x[i];
}

__global__ void copy_f32(const float* __restrict__ x, float* __restrict__ y, int n) {
  int i = blockIdx.x * 256 + threadIdx.x;
  if (i < n) y[i] = x[i];
}

// ew1 [E,H,F] -> f16 [E,F,H]
__global__ void tr_ew1(const float* __restrict__ x, _Float16* __restrict__ y, int n) {
  int i = blockIdx.x * 256 + threadIdx.x;
  if (i >= n) return;
  int hh = i % HDIM, rest = i / HDIM, f = rest % FDIM, e = rest / FDIM;
  y[i] = (_Float16)x[(size_t)(e * HDIM + hh) * FDIM + f];
}

// ew2 [E,F,H] -> f16 [E,H,F]
__global__ void tr_ew2(const float* __restrict__ x, _Float16* __restrict__ y, int n) {
  int i = blockIdx.x * 256 + threadIdx.x;
  if (i >= n) return;
  int f = i % FDIM, rest = i / FDIM, hh = rest % HDIM, e = rest / HDIM;
  y[i] = (_Float16)x[(size_t)(e * FDIM + f) * HDIM + hh];
}

__global__ void mean_rows(const float* __restrict__ x, float* __restrict__ y, int T) {
  int i = blockIdx.x * 256 + threadIdx.x;   // B*H total
  int b = i / HDIM, hh = i % HDIM;
  float s = 0.f;
  for (int tt = 0; tt < T; tt++) s += x[((size_t)b * T + tt) * HDIM + hh];
  y[i] = s / (float)T;
}

// probs = softmax(concat(x, ctx[b]) @ W^T + b) over NE experts
__global__ void gate_softmax(const float* __restrict__ x, const float* __restrict__ ctx,
                             const float* __restrict__ w, const float* __restrict__ bb,
                             float* __restrict__ probs, int T) {
  int i = blockIdx.x * 256 + threadIdx.x;
  if (i >= BQ * T) return;
  int b = i / T;
  float lg[NE];
  float mx = -3.0e38f;
  const float* xr = x + (size_t)i * HDIM;
  const float* cb = ctx + b * HDIM;
  for (int e = 0; e < NE; e++) {
    const float* we = w + e * 2 * HDIM;
    float a = bb[e];
    for (int k = 0; k < HDIM; k++) a += xr[k] * we[k];
    for (int k = 0; k < HDIM; k++) a += cb[k] * we[HDIM + k];
    lg[e] = a; mx = fmaxf(mx, a);
  }
  float s = 0.f;
  for (int e = 0; e < NE; e++) { lg[e] = __expf(lg[e] - mx); s += lg[e]; }
  for (int e = 0; e < NE; e++) probs[(size_t)i * NE + e] = lg[e] / s;
}

// ---------------------------------------------------------------------------
extern "C" void kernel_launch(void* const* d_in, const int* in_sizes, int n_in,
                              void* d_out, int out_size, void* d_ws, size_t ws_size,
                              hipStream_t stream) {
  const float* query = (const float*)d_in[0];
  const float* image = (const float*)d_in[1];
  const float* text  = (const float*)d_in[2];
  const float* sa_in_w  = (const float*)d_in[3];
  const float* sa_in_b  = (const float*)d_in[4];
  const float* sa_out_w = (const float*)d_in[5];
  const float* sa_out_b = (const float*)d_in[6];
  const float* ca_in_w  = (const float*)d_in[7];
  const float* ca_in_b  = (const float*)d_in[8];
  const float* ca_out_w = (const float*)d_in[9];
  const float* ca_out_b = (const float*)d_in[10];
  const float* img_gate_w = (const float*)d_in[11];
  const float* img_gate_b = (const float*)d_in[12];
  const float* txt_gate_w = (const float*)d_in[13];
  const float* txt_gate_b = (const float*)d_in[14];
  const float* lnq_g = (const float*)d_in[15];
  const float* lnq_b = (const float*)d_in[16];
  const float* lnc_g = (const float*)d_in[17];
  const float* lnc_b = (const float*)d_in[18];
  const float* lnf_g = (const float*)d_in[19];
  const float* lnf_b = (const float*)d_in[20];
  const float* ew1 = (const float*)d_in[21];
  const float* eb1 = (const float*)d_in[22];
  const float* ew2 = (const float*)d_in[23];
  const float* eb2 = (const float*)d_in[24];

  const int M = BQ * TLEN;           // 2048 tokens per stream
  char* w = (char*)d_ws;
  auto alloc = [&](size_t bytes) { void* p = (void*)w; w += (bytes + 255) & ~(size_t)255; return p; };

  _Float16* w_sa_in  = (_Float16*)alloc((size_t)3 * HDIM * HDIM * 2);
  _Float16* w_sa_out = (_Float16*)alloc((size_t)HDIM * HDIM * 2);
  _Float16* w_ca_in  = (_Float16*)alloc((size_t)3 * HDIM * HDIM * 2);
  _Float16* w_ca_out = (_Float16*)alloc((size_t)HDIM * HDIM * 2);
  _Float16* ew1T = (_Float16*)alloc((size_t)NE * FDIM * HDIM * 2);
  _Float16* ew2T = (_Float16*)alloc((size_t)NE * HDIM * FDIM * 2);
  _Float16* xn     = (_Float16*)alloc((size_t)M * HDIM * 2);
  _Float16* x_img  = (_Float16*)alloc((size_t)M * HDIM * 2);
  _Float16* xf     = (_Float16*)alloc((size_t)M * HDIM * 2);
  _Float16* attn   = (_Float16*)alloc((size_t)M * HDIM * 2);
  _Float16* qkv_a  = (_Float16*)alloc((size_t)M * 3 * HDIM * 2);
  _Float16* qkv_b  = (_Float16*)alloc((size_t)M * 3 * HDIM * 2);
  _Float16* qkv_c  = (_Float16*)alloc((size_t)M * 3 * HDIM * 2);
  _Float16* hbuf   = (_Float16*)alloc((size_t)M * FDIM * 2);
  float* Q1 = (float*)alloc((size_t)M * HDIM * 4);
  float* Q2 = (float*)alloc((size_t)M * HDIM * 4);
  float* img_ctx = (float*)alloc((size_t)BQ * HDIM * 4);
  float* txt_ctx = (float*)alloc((size_t)BQ * HDIM * 4);
  float* probs_img = (float*)alloc((size_t)M * NE * 4);
  float* probs_txt = (float*)alloc((size_t)M * NE * 4);

  float* outQ = (float*)d_out;
  float* outI = outQ + (size_t)M * HDIM;

  // --- weight conversion / transposition (f32 -> f16) ---
  int nw;
  nw = 3 * HDIM * HDIM; cvt_f16<<<(nw + 255) / 256, 256, 0, stream>>>(sa_in_w, w_sa_in, nw);
  nw = HDIM * HDIM;     cvt_f16<<<(nw + 255) / 256, 256, 0, stream>>>(sa_out_w, w_sa_out, nw);
  nw = 3 * HDIM * HDIM; cvt_f16<<<(nw + 255) / 256, 256, 0, stream>>>(ca_in_w, w_ca_in, nw);
  nw = HDIM * HDIM;     cvt_f16<<<(nw + 255) / 256, 256, 0, stream>>>(ca_out_w, w_ca_out, nw);
  nw = NE * HDIM * FDIM;
  tr_ew1<<<(nw + 255) / 256, 256, 0, stream>>>(ew1, ew1T, nw);
  tr_ew2<<<(nw + 255) / 256, 256, 0, stream>>>(ew2, ew2T, nw);

  // --- self attention ---
  ln_f16<<<M, 256, 0, stream>>>(query, lnq_g, lnq_b, xn, HDIM);
  gemm_wmma<0><<<dim3(M / 128, 3 * HDIM / 128), 256, 0, stream>>>(
      xn, w_sa_in, sa_in_b, nullptr, nullptr, qkv_a, M, 3 * HDIM, HDIM, 0);
  attn_wmma<<<BQ * NHH * (TLEN / 128), 256, 0, stream>>>(qkv_a, qkv_a, attn, TLEN, TLEN);
  gemm_wmma<1><<<dim3(M / 128, HDIM / 128), 256, 0, stream>>>(
      attn, w_sa_out, sa_out_b, query, Q1, nullptr, M, HDIM, HDIM, 0);

  // --- cross attention (q from Q1 normed, k/v from image tokens) ---
  ln_f16<<<M, 256, 0, stream>>>(Q1, lnc_g, lnc_b, xn, HDIM);
  cvt_f16<<<(M * HDIM + 255) / 256, 256, 0, stream>>>(image, x_img, M * HDIM);
  gemm_wmma<0><<<dim3(M / 128, 3 * HDIM / 128), 256, 0, stream>>>(
      xn, w_ca_in, ca_in_b, nullptr, nullptr, qkv_b, M, 3 * HDIM, HDIM, 0);
  gemm_wmma<0><<<dim3(M / 128, 3 * HDIM / 128), 256, 0, stream>>>(
      x_img, w_ca_in, ca_in_b, nullptr, nullptr, qkv_c, M, 3 * HDIM, HDIM, 0);
  attn_wmma<<<BQ * NHH * (TLEN / 128), 256, 0, stream>>>(qkv_b, qkv_c, attn, TLEN, TLEN);
  gemm_wmma<1><<<dim3(M / 128, HDIM / 128), 256, 0, stream>>>(
      attn, w_ca_out, ca_out_b, Q1, Q2, nullptr, M, HDIM, HDIM, 0);

  // --- gating ---
  mean_rows<<<(BQ * HDIM) / 256, 256, 0, stream>>>(image, img_ctx, TLEN);
  mean_rows<<<(BQ * HDIM) / 256, 256, 0, stream>>>(text, txt_ctx, TLEN);
  gate_softmax<<<(M + 255) / 256, 256, 0, stream>>>(image, txt_ctx, img_gate_w, img_gate_b, probs_img, TLEN);
  gate_softmax<<<(M + 255) / 256, 256, 0, stream>>>(Q2, img_ctx, txt_gate_w, txt_gate_b, probs_txt, TLEN);

  // --- MoE: out = residual + sum_e prob_e * (GELU(x W1_e + b1) W2_e + b2) ---
  ln_f16<<<M, 256, 0, stream>>>(Q2, lnf_g, lnf_b, xf, HDIM);
  copy_f32<<<(M * HDIM + 255) / 256, 256, 0, stream>>>(Q2, outQ, M * HDIM);
  copy_f32<<<(M * HDIM + 255) / 256, 256, 0, stream>>>(image, outI, M * HDIM);
  for (int e = 0; e < NE; e++) {
    const _Float16* w1e = ew1T + (size_t)e * FDIM * HDIM;
    const _Float16* w2e = ew2T + (size_t)e * HDIM * FDIM;
    const float* b1e = eb1 + e * FDIM;
    const float* b2e = eb2 + e * HDIM;
    // query stream
    gemm_wmma<2><<<dim3(M / 128, FDIM / 128), 256, 0, stream>>>(
        xf, w1e, b1e, nullptr, nullptr, hbuf, M, FDIM, HDIM, 0);
    gemm_wmma<3><<<dim3(M / 128, HDIM / 128), 256, 0, stream>>>(
        hbuf, w2e, b2e, probs_txt, outQ, nullptr, M, HDIM, FDIM, e);
    // image stream (no LN)
    gemm_wmma<2><<<dim3(M / 128, FDIM / 128), 256, 0, stream>>>(
        x_img, w1e, b1e, nullptr, nullptr, hbuf, M, FDIM, HDIM, 0);
    gemm_wmma<3><<<dim3(M / 128, HDIM / 128), 256, 0, stream>>>(
        hbuf, w2e, b2e, probs_img, outI, nullptr, M, HDIM, FDIM, e);
  }
  (void)in_sizes; (void)n_in; (void)out_size; (void)ws_size;
}
